// MlpMixer_56779467653701
// MI455X (gfx1250) — matrix operand order repaired and checked by reading, not compiled, not run on hardware
//
#include <hip/hip_runtime.h>
#include <hip/hip_bf16.h>

// Spiking MLP-Mixer for MI455X (gfx1250).
// - All GEMMs via v_wmma_f32_16x16x32_f16 (f32 accumulate).
// - Activations staged in LDS in WMMA A-fragment layout (one 32B ds load/lane).
// - +-1 weights pre-swizzled into WMMA B-fragment layout in workspace.
// - LIF membrane state V kept in fragment-major layout -> float4 coalesced RMW.
// - Each wave computes 2 N-tiles per A-fragment load (A reuse).
// - K-loop partially unrolled (x2) to bound VGPR pressure: round-2 full unroll
//   made the compiler prefetch 128 VGPRs of weights and spill to scratch.
// - Block-outer / time-inner loop: only one block's V live (25MB, ws ~55MB).
// - 2 x 96KB LDS A-buffers: uses CDNA5's 320KB WGP LDS. 512 thr = 16 wave32.

static constexpr int Bb    = 32;
static constexpr int Tt    = 256;
static constexpr int Cc    = 192;
static constexpr int NBLK  = 12;
static constexpr int TSEQ  = 4;
static constexpr int NCLS  = 1000;
static constexpr int WAVES = 16;     // 512 threads

typedef __attribute__((ext_vector_type(16))) _Float16 v16h;
typedef __attribute__((ext_vector_type(8)))  float    v8f;

// ---- WMMA fragment layouts (CDNA5 ISA §7.12.2) -----------------
// A (16x32 f16): lane = m%16 + 16*((k>>3)&1); slot = (k&7) + 8*(k>>4)
__device__ __forceinline__ int a_off(int m, int k, int KT) {
  int mt = m >> 4, kt = k >> 5, mm = m & 15, kk = k & 31;
  int half = (kk >> 3) & 1;
  int lane = mm + (half << 4);
  int slot = (kk & 7) + ((kk >> 4) << 3);
  return (((mt * KT + kt) << 5) + lane) * 16 + slot;
}
// B (32x16 f16): lane = n%16 + 16*((k>>3)&1); slot = (k&7) + 8*(k>>4)
__device__ __forceinline__ int b_off(int k, int n, int KT) {
  int nt = n >> 4, kt = k >> 5, kk = k & 31, nn = n & 15;
  int half = (kk >> 3) & 1;
  int lane = nn + (half << 4);
  int slot = (kk & 7) + ((kk >> 4) << 3);
  return (((nt * KT + kt) << 5) + lane) * 16 + slot;
}

// ---- weight conversion: bool -> +-1 f16 in B-fragment layout ---------------
__global__ void conv_tok(const unsigned char* __restrict__ tk,
                         _Float16* __restrict__ w) {
  int i = blockIdx.x * 256 + threadIdx.x;          // 24*65536 elements
  int mat = i >> 16;
  int r = i & 65535;
  int k = r >> 8, n = r & 255;
  _Float16 v = tk[i] ? (_Float16)1.f : (_Float16)-1.f;
  w[(mat << 16) + b_off(k, n, 8)] = v;
}
__global__ void conv_ch(const unsigned char* __restrict__ ck,
                        _Float16* __restrict__ w) {
  int i = blockIdx.x * 256 + threadIdx.x;          // 24*36864 elements
  int mat = i / 36864;
  int r = i - mat * 36864;
  int k = r / 192, n = r - k * 192;
  _Float16 v = ck[i] ? (_Float16)1.f : (_Float16)-1.f;
  w[mat * 36864 + b_off(k, n, 6)] = v;
}

// ---- patchify: x(B,3,128,128) -> xseq[t][b][T][C], replicated over t -------
__global__ void patchify(const float* __restrict__ x, float* __restrict__ xseq) {
  int i = blockIdx.x * 256 + threadIdx.x;          // B*T*C elements
  int c = i % Cc; int rem = i / Cc;
  int t = rem % Tt; int b = rem / Tt;
  int ch = c % 3; int pp = c / 3; int ix = pp & 7; int iy = pp >> 3;
  int px = t & 15, py = t >> 4;
  float v = x[((size_t)b * 3 + ch) * (128 * 128) + (py * 8 + iy) * 128 + (px * 8 + ix)];
  for (int ts = 0; ts < TSEQ; ++ts)
    xseq[(((size_t)ts * Bb + b) * Tt + t) * Cc + c] = v;
}

// ---- LN over channels, staged to LDS in A-fragment layout ------------------
// 512 threads: 2 threads per token row, partial sums combined via shfl_xor.
__device__ __forceinline__ void ln_stage(const float* __restrict__ xb,
                                         const float* __restrict__ s,
                                         const float* __restrict__ bsc,
                                         _Float16* __restrict__ abuf,
                                         int tokenMode) {
  int r = threadIdx.x >> 1;
  int h = threadIdx.x & 1;
  const float* row = xb + r * Cc;
  int c0 = h * (Cc / 2);
  float su = 0.f, sq = 0.f;
  for (int c = c0; c < c0 + Cc / 2; ++c) { float v = row[c]; su += v; sq += v * v; }
  su += __shfl_xor(su, 1);
  sq += __shfl_xor(sq, 1);
  float mu  = su * (1.f / Cc);
  float var = sq * (1.f / Cc) - mu * mu;
  float rs  = rsqrtf(var + 1e-6f);
  for (int c = c0; c < c0 + Cc / 2; ++c) {
    float y = (row[c] - mu) * rs * s[c] + bsc[c];
    if (tokenMode) abuf[a_off(c, r, 8)] = (_Float16)y;   // A = [C][T]
    else           abuf[a_off(r, c, 6)] = (_Float16)y;   // A = [T][C]
  }
}

// ---- WMMA GEMM + LIF. POST 0: spike->LDS (next A). POST 1: x += spike. -----
// Each wave computes 2 N-tiles per A-fragment load. V is fragment-major:
// vbuf[((tile)*32 + lane)*8 + d], tile = mt*nNt + nt.
template <int M, int K, int N, int POST>
__device__ __forceinline__ void gemm_lif(const _Float16* __restrict__ alds,
                                         const _Float16* __restrict__ wg,
                                         float scale, float* __restrict__ vbuf,
                                         int vinit, _Float16* __restrict__ sout,
                                         float* __restrict__ xb, int sm, int sn) {
  constexpr int nMt = M / 16, nNt = N / 16, nKt = K / 32, KTn = N / 32;
  constexpr int nPair = nNt / 2;
  int wave = threadIdx.x >> 5;
  int lane = threadIdx.x & 31;
  for (int p = wave; p < nMt * nPair; p += WAVES) {
    int mt = p / nPair, nt0 = (p - mt * nPair) * 2;
    v8f acc0 = {}, acc1 = {};
    const _Float16* ap  = alds + ((size_t)(mt * nKt) * 32 + lane) * 16;
    const _Float16* wp0 = wg   + ((size_t)(nt0 * nKt) * 32 + lane) * 16;
    const _Float16* wp1 = wp0  + (size_t)nKt * 512;
    // Partial unroll: keep live set ~(2 A + 4 W + 2 acc) fragments -> no spill.
#pragma unroll 2
    for (int kt = 0; kt < nKt; ++kt) {
      v16h a  = *(const v16h*)(ap  + kt * 512);
      v16h w0 = *(const v16h*)(wp0 + kt * 512);
      v16h w1 = *(const v16h*)(wp1 + kt * 512);
      acc0 = __builtin_amdgcn_wmma_f32_16x16x32_f16(false, a, false, w0,
                                                    (short)0, acc0, false, false);
      acc1 = __builtin_amdgcn_wmma_f32_16x16x32_f16(false, a, false, w1,
                                                    (short)0, acc1, false, false);
    }
    int mm = (lane >> 4) << 3;          // 0 or 8
    int mB = mm + mt * 16;
#pragma unroll
    for (int sub = 0; sub < 2; ++sub) {
      v8f acc = sub ? acc1 : acc0;
      int nt = nt0 + sub;
      int nn = (lane & 15) + nt * 16;
      float* vp = vbuf + ((size_t)(mt * nNt + nt) * 32 + lane) * 8;
      float4 vA, vB;
      if (vinit) { vA = float4{0, 0, 0, 0}; vB = float4{0, 0, 0, 0}; }
      else       { vA = ((const float4*)vp)[0]; vB = ((const float4*)vp)[1]; }
      float vv[8] = {vA.x, vA.y, vA.z, vA.w, vB.x, vB.y, vB.z, vB.w};
      float ss[8];
#pragma unroll
      for (int d = 0; d < 8; ++d) {
        float z  = scale * acc[d];
        float Vn = 0.5f * vv[d] + z;
        bool spk = Vn > 1.f;
        ss[d] = spk ? 1.f : 0.f;
        vv[d] = spk ? 0.f : Vn;
      }
      ((float4*)vp)[0] = float4{vv[0], vv[1], vv[2], vv[3]};
      ((float4*)vp)[1] = float4{vv[4], vv[5], vv[6], vv[7]};
      if (POST == 0) {
        // spike -> next A-fragment. For fixed lane: slot fixed, lane steps with d.
        int kk   = nn & 31;
        int hh   = (kk >> 3) & 1;
        int slot = (kk & 7) + ((kk >> 4) << 3);
        int base = (((mt * KTn + (nn >> 5)) << 5) + (mm + (hh << 4))) * 16 + slot;
#pragma unroll
        for (int d = 0; d < 8; ++d) sout[base + d * 16] = (_Float16)ss[d];
      } else {
        int base = mB * sm + nn * sn;
#pragma unroll
        for (int d = 0; d < 8; ++d)
          if (ss[d] != 0.f) xb[base + d * sm] += 1.f;
      }
    }
  }
}

// ---- main mixer: 1 workgroup per batch element, block-outer / time-inner ---
__global__ void __launch_bounds__(512) mixer_main(
    float* __restrict__ xseq, const _Float16* __restrict__ tokW,
    const _Float16* __restrict__ chW, float* __restrict__ V,
    float* __restrict__ pooled, const float* __restrict__ ln_s,
    const float* __restrict__ ln_b, const float* __restrict__ fs,
    const float* __restrict__ fb, const float* __restrict__ tsc,
    const float* __restrict__ csc) {
  __shared__ __align__(32) _Float16 abufA[Tt * Cc];   // 96 KB
  __shared__ __align__(32) _Float16 abufB[Tt * Cc];   // 96 KB
  __shared__ float spool[Cc];
  int b = blockIdx.x;
  float* v0 = V + ((size_t)0 * Bb + b) * 49152;
  float* v1 = V + ((size_t)1 * Bb + b) * 49152;
  float* v2 = V + ((size_t)2 * Bb + b) * 49152;
  float* v3 = V + ((size_t)3 * Bb + b) * 49152;
  for (int blk = 0; blk < NBLK; ++blk) {
    const float* lns = ln_s + blk * Cc;
    const float* lnb = ln_b + blk * Cc;
    float t0s = tsc[blk * 2 + 0], t1s = tsc[blk * 2 + 1];
    float c0s = csc[blk * 2 + 0], c1s = csc[blk * 2 + 1];
    const _Float16* w0 = tokW + (size_t)(blk * 2 + 0) * 65536;
    const _Float16* w1 = tokW + (size_t)(blk * 2 + 1) * 65536;
    const _Float16* w2 = chW  + (size_t)(blk * 2 + 0) * 36864;
    const _Float16* w3 = chW  + (size_t)(blk * 2 + 1) * 36864;
    for (int t = 0; t < TSEQ; ++t) {
      float* xb = xseq + ((size_t)t * Bb + b) * Tt * Cc;
      int vin = (t == 0);
      // token mixing: A = LN(x)^T [C=192, T=256] @ [256,256]
      ln_stage(xb, lns, lnb, abufA, 1);
      __syncthreads();
      gemm_lif<192, 256, 256, 0>(abufA, w0, t0s, v0, vin, abufB, nullptr, 0, 0);
      __syncthreads();
      gemm_lif<192, 256, 256, 1>(abufB, w1, t1s, v1, vin, nullptr, xb, 1, Cc);
      __threadfence_block();
      __syncthreads();
      // channel mixing: A = LN(x) [T=256, C=192] @ [192,192]
      ln_stage(xb, lns, lnb, abufA, 0);
      __syncthreads();
      gemm_lif<256, 192, 192, 0>(abufA, w2, c0s, v2, vin, abufB, nullptr, 0, 0);
      __syncthreads();
      gemm_lif<256, 192, 192, 1>(abufB, w3, c1s, v3, vin, nullptr, xb, Cc, 1);
      __threadfence_block();
      __syncthreads();
    }
  }
  // final LN + token-mean pooling per timestep
  for (int t = 0; t < TSEQ; ++t) {
    const float* xb = xseq + ((size_t)t * Bb + b) * Tt * Cc;
    for (int i = threadIdx.x; i < Cc; i += 512) spool[i] = 0.f;
    __syncthreads();
    {
      int r = threadIdx.x >> 1;
      int h = threadIdx.x & 1;
      const float* row = xb + r * Cc;
      int c0 = h * (Cc / 2);
      float su = 0.f, sq = 0.f;
      for (int c = c0; c < c0 + Cc / 2; ++c) { float v = row[c]; su += v; sq += v * v; }
      su += __shfl_xor(su, 1);
      sq += __shfl_xor(sq, 1);
      float mu  = su * (1.f / Cc);
      float var = sq * (1.f / Cc) - mu * mu;
      float rs  = rsqrtf(var + 1e-6f);
      for (int c = c0; c < c0 + Cc / 2; ++c) {
        float y = (row[c] - mu) * rs * fs[c] + fb[c];
        atomicAdd(&spool[c], y * (1.f / Tt));
      }
    }
    __syncthreads();
    for (int i = threadIdx.x; i < Cc; i += 512)
      pooled[((size_t)t * Bb + b) * Cc + i] = spool[i];
    __syncthreads();
  }
}

// ---- head: out[b][n] = hs * sum_t sum_c pooled[t][b][c] * sign(hk[c][n]) ---
__global__ void head_kernel(const float* __restrict__ pooled,
                            const unsigned char* __restrict__ hk,
                            const float* __restrict__ hs,
                            float* __restrict__ out) {
  int n = blockIdx.x * 256 + threadIdx.x;
  int b = blockIdx.y;
  if (n >= NCLS) return;
  float acc = 0.f;
  for (int c = 0; c < Cc; ++c) {
    float ps = 0.f;
    for (int t = 0; t < TSEQ; ++t) ps += pooled[((size_t)t * Bb + b) * Cc + c];
    acc += hk[(size_t)c * NCLS + n] ? ps : -ps;
  }
  out[(size_t)b * NCLS + n] = acc * hs[0];
}

extern "C" void kernel_launch(void* const* d_in, const int* in_sizes, int n_in,
                              void* d_out, int out_size, void* d_ws, size_t ws_size,
                              hipStream_t stream) {
  const float*         x      = (const float*)d_in[0];
  const unsigned char* tok_k  = (const unsigned char*)d_in[1];
  const unsigned char* ch_k   = (const unsigned char*)d_in[2];
  const unsigned char* head_k = (const unsigned char*)d_in[3];
  const float*         tok_s  = (const float*)d_in[4];
  const float*         ch_s   = (const float*)d_in[5];
  const float*         head_s = (const float*)d_in[6];
  const float*         ln_s   = (const float*)d_in[7];
  const float*         ln_b   = (const float*)d_in[8];
  const float*         fin_s  = (const float*)d_in[9];
  const float*         fin_b  = (const float*)d_in[10];

  char* ws = (char*)d_ws;
  _Float16* tokW   = (_Float16*)(ws);                 // 24*65536 f16 = 3,145,728 B
  _Float16* chW    = (_Float16*)(ws + 3145728);       // 24*36864 f16 = 1,769,472 B
  float*    xseq   = (float*)(ws + 4915200);          // 4*32*256*192 f32 = 25,165,824 B
  float*    V      = (float*)(ws + 30081024);         // 4*32*49152  f32 = 25,165,824 B
  float*    pooled = (float*)(ws + 55246848);         // 4*32*192    f32 = 98,304 B

  conv_tok<<<24 * 65536 / 256, 256, 0, stream>>>(tok_k, tokW);
  conv_ch<<<24 * 36864 / 256, 256, 0, stream>>>(ch_k, chW);
  patchify<<<Bb * Tt * Cc / 256, 256, 0, stream>>>(x, xseq);
  mixer_main<<<Bb, 512, 0, stream>>>(xseq, tokW, chW, V, pooled,
                                     ln_s, ln_b, fin_s, fin_b, tok_s, ch_s);
  head_kernel<<<dim3((NCLS + 255) / 256, Bb), 256, 0, stream>>>(pooled, head_k,
                                                                head_s, (float*)d_out);
}
